// Attention_111669149705
// MI455X (gfx1250) — matrix operand order repaired and checked by reading, not compiled, not run on hardware
//
#include <hip/hip_runtime.h>

typedef __attribute__((ext_vector_type(16))) __bf16 v16bf;
typedef __attribute__((ext_vector_type(8)))  float  v8f;
typedef unsigned int  u32x4 __attribute__((ext_vector_type(4)));
typedef int           i32x8 __attribute__((ext_vector_type(8)));
typedef int           i32x4 __attribute__((ext_vector_type(4)));
typedef unsigned short ushort_t;
typedef unsigned int   uint32;

#define B_   4
#define S_   2048
#define D_   768
#define H_   12
#define DH_  64
#define BS_  (B_ * S_)

#if __has_builtin(__builtin_amdgcn_tensor_load_to_lds) && __has_builtin(__builtin_amdgcn_s_wait_tensorcnt)
#define USE_TDM 1
#else
#define USE_TDM 0
#endif

__device__ __forceinline__ ushort_t f2bf(float f) {
  union { float f; uint32 u; } v; v.f = f;
  uint32 u = v.u;
  uint32 r = u + 0x7FFFu + ((u >> 16) & 1u);   // round-to-nearest-even
  return (ushort_t)(r >> 16);
}

union FragAB { uint32 u[8]; v16bf v; };
union FragC  { float  f[8]; v8f  v; };

// ---------------------------------------------------------------- pack kernels
__global__ void k_f32_to_bf16(const float* __restrict__ in,
                              ushort_t* __restrict__ out, int n) {
  int i = blockIdx.x * blockDim.x + threadIdx.x;
  if (i < n) out[i] = f2bf(in[i]);
}

// W[h, d, e] -> Wt[h*64+e][d]   (transposed [N,K] so B tiles are contiguous)
__global__ void k_pack_qkv(const float* __restrict__ w, ushort_t* __restrict__ out) {
  int t = blockIdx.x * blockDim.x + threadIdx.x;
  if (t >= D_ * D_) return;
  int n = t / D_, d = t % D_;
  int h = n >> 6, e = n & 63;
  out[t] = f2bf(w[((size_t)h * D_ + d) * DH_ + e]);
}

// W_O flat [h*64+e][d] -> Wt[d][h*64+e]   (plain transpose to [N,K])
__global__ void k_pack_oT(const float* __restrict__ w, ushort_t* __restrict__ out) {
  int t = blockIdx.x * blockDim.x + threadIdx.x;
  if (t >= D_ * D_) return;
  int n = t / D_, k = t % D_;
  out[t] = f2bf(w[(size_t)k * D_ + n]);
}

// ---------------------------------------------------------------- WMMA GEMM
// C[8192,768] = A[8192,768] * Bt[768(N),768(K)]^T + bias
// mode 0: out bf16 scattered to [B,H,S,Dh]   (Q, K)
// mode 1: out bf16 scattered to [B,H,Dh,S]   (V transposed)
// mode 2: out fp32 row-major [8192,768]      (final projection)
#define GT_M 128
#define GT_N 64
#define GT_K 32
#define LDT  40   // bf16 elements per LDS tile row (32 data + 8 pad -> 80 B, 16B-aligned)
#define NT_  (D_ / GT_K)

// Generic 2-D TDM: DMA tileRows x 32 bf16 from a row-major [dim1, 768] tensor
// into LDS, padding each 64 B row with 16 B so rows land on the LDT stride.
__device__ __forceinline__ void tdm_issue_2d(const ushort_t* gptr, ushort_t* ldsDst,
                                             int tileRows, int dim1) {
#if USE_TDM
  unsigned int ldsAddr = (unsigned int)(size_t)ldsDst;   // low 32 bits = LDS byte offset
  unsigned long long ga = (unsigned long long)(size_t)gptr;
  u32x4 g0;
  g0[0] = 1u;                                            // count=1, load, user mode
  g0[1] = ldsAddr;                                       // lds_addr
  g0[2] = (uint32)ga;                                    // global_addr[31:0]
  g0[3] = (uint32)((ga >> 32) & 0x01FFFFFFull) | (2u << 30); // ga[56:32] | type=2
  i32x8 g1;
  g1[0] = (int)((1u << 16) |                             // data_size = 2 bytes
                (1u << 20) |                             // pad_enable
                (3u << 22) |                             // pad_interval: 16 DWORDs (64 B)
                (3u << 25));                             // pad_amount:   4 DWORDs (16 B)
  g1[1] = (int)(((uint32)D_ & 0xFFFFu) << 16);           // tensor_dim0[15:0] @ bits 63:48
  g1[2] = (int)((((uint32)D_) >> 16) | (((uint32)dim1 & 0xFFFFu) << 16)); // dim0 hi | dim1 lo
  g1[3] = (int)((((uint32)dim1) >> 16) | ((uint32)GT_K << 16));           // dim1 hi | tile_dim0
  g1[4] = (int)tileRows;                                 // tile_dim1, tile_dim2=0
  g1[5] = (int)D_;                                       // tensor_dim0_stride[31:0]
  g1[6] = 0;                                             // stride hi | dim1_stride lo
  g1[7] = 0;
  i32x4 zz = {0, 0, 0, 0};
  i32x8 z8 = {0, 0, 0, 0, 0, 0, 0, 0};
  __builtin_amdgcn_tensor_load_to_lds(g0, g1, zz, zz, z8, 0);
#endif
}

__global__ __launch_bounds__(256)
void k_gemm_bf16(const ushort_t* __restrict__ A, const ushort_t* __restrict__ Bt,
                 const float* __restrict__ bias, void* __restrict__ outp, int mode) {
  __shared__ __align__(16) ushort_t As[2][GT_M * LDT];
  __shared__ __align__(16) ushort_t Bs[2][GT_N * LDT];

  const int tid  = threadIdx.x;
  const int lane = tid & 31;
  const int w    = tid >> 5;      // wave 0..7
  const int wm   = w & 3;         // 4 waves along M (32 rows each)
  const int wn   = w >> 2;        // 2 waves along N (32 cols each)
  const int hi   = (lane >> 4) & 1;
  const int l16  = lane & 15;
  const int m0   = blockIdx.x * GT_M;
  const int n0   = blockIdx.y * GT_N;

  // stage helpers: A tile = 128x32 bf16 of X; B tile = 64x32 bf16 of Wt -----
  auto stage = [&](int kt, int buf) {
#if USE_TDM
    if (w == 0) tdm_issue_2d(A  + (size_t)m0 * D_ + kt * GT_K, &As[buf][0], GT_M, BS_);
    if (w == 1) tdm_issue_2d(Bt + (size_t)n0 * D_ + kt * GT_K, &Bs[buf][0], GT_N, D_);
#else
    #pragma unroll
    for (int i = 0; i < 2; ++i) {
      int c   = tid + i * 256;
      int row = c >> 2, cp = c & 3;
      const uint4* src = reinterpret_cast<const uint4*>(
          A + (size_t)(m0 + row) * D_ + kt * GT_K + cp * 8);
      *reinterpret_cast<uint4*>(&As[buf][row * LDT + cp * 8]) = *src;
    }
    {
      int row = tid >> 2, cp = tid & 3;
      const uint4* src = reinterpret_cast<const uint4*>(
          Bt + (size_t)(n0 + row) * D_ + kt * GT_K + cp * 8);
      *reinterpret_cast<uint4*>(&Bs[buf][row * LDT + cp * 8]) = *src;
    }
#endif
  };
  auto waitStage = [&]() {
#if USE_TDM
    if (w <= 1) __builtin_amdgcn_s_wait_tensorcnt(0);
#endif
  };

  v8f zero8 = {0.f,0.f,0.f,0.f,0.f,0.f,0.f,0.f};
  v8f acc[2][2] = { { zero8, zero8 }, { zero8, zero8 } };

  // prologue: fill buffer 0
  stage(0, 0);
  waitStage();
  __syncthreads();

  for (int kt = 0; kt < NT_; ++kt) {
    const int cur = kt & 1, nxt = cur ^ 1;
    if (kt + 1 < NT_) stage(kt + 1, nxt);   // async DMA into other buffer

    FragAB af[2], bfr[2];
    #pragma unroll
    for (int ms = 0; ms < 2; ++ms) {
      int row = wm * 32 + ms * 16 + l16;
      #pragma unroll
      for (int r = 0; r < 8; ++r) {
        int k0 = hi * 16 + 2 * r;   // A-layout K pairs
        af[ms].u[r] = *reinterpret_cast<const uint32*>(&As[cur][row * LDT + k0]);
      }
    }
    #pragma unroll
    for (int ns = 0; ns < 2; ++ns) {
      int n = wn * 32 + ns * 16 + l16;
      #pragma unroll
      for (int r = 0; r < 8; ++r) {
        int k0 = hi * 16 + 2 * r;                               // B-layout K pairs
        bfr[ns].u[r] = *reinterpret_cast<const uint32*>(&Bs[cur][n * LDT + k0]);
      }
    }
    #pragma unroll
    for (int ms = 0; ms < 2; ++ms)
      #pragma unroll
      for (int ns = 0; ns < 2; ++ns)
        acc[ms][ns] = __builtin_amdgcn_wmma_f32_16x16x32_bf16(
            false, af[ms].v, false, bfr[ns].v, (short)0, acc[ms][ns], false, false);

    if (kt + 1 < NT_) {
      waitStage();
      __syncthreads();
    }
  }

  // epilogue
  #pragma unroll
  for (int ms = 0; ms < 2; ++ms) {
    #pragma unroll
    for (int ns = 0; ns < 2; ++ns) {
      FragC c; c.v = acc[ms][ns];
      int n = n0 + wn * 32 + ns * 16 + l16;
      float bn = bias[n];
      #pragma unroll
      for (int i = 0; i < 8; ++i) {
        int m = m0 + wm * 32 + ms * 16 + i + hi * 8;   // C-layout rows
        float val = c.f[i] + bn;
        if (mode == 2) {
          ((float*)outp)[(size_t)m * D_ + n] = val;
        } else {
          int b = m >> 11, s = m & (S_ - 1);
          int h = n >> 6,  e = n & 63;
          ushort_t bv = f2bf(val);
          if (mode == 0)
            ((ushort_t*)outp)[((size_t)(b * H_ + h) * S_ + s) * DH_ + e] = bv;
          else
            ((ushort_t*)outp)[((size_t)(b * H_ + h) * DH_ + e) * S_ + s] = bv;
        }
      }
    }
  }
}

// ---------------------------------------------------------------- flash attention
// Q,K: bf16 [B,H,S,64]; Vt: bf16 [B,H,64,S]; Z out: bf16 [B*S, 768]
// Softmax tracked in base-2 domain: scale = (1/sqrt(64)) * log2(e).
#define SCALE_LOG2E 0.180336880f

__global__ __launch_bounds__(256)
void k_attn(const ushort_t* __restrict__ Q, const ushort_t* __restrict__ K,
            const ushort_t* __restrict__ Vt, ushort_t* __restrict__ Z) {
  __shared__ __align__(16) ushort_t Plds[8][16 * 34];   // per-wave P transpose scratch

  const int tid  = threadIdx.x;
  const int lane = tid & 31;
  const int w    = tid >> 5;
  const int hi   = (lane >> 4) & 1;
  const int l16  = lane & 15;
  const int qt   = blockIdx.x;      // 16 q-tiles of 128
  const int h    = blockIdx.y;
  const int b    = blockIdx.z;
  const int qr0  = qt * 128 + w * 16;   // this wave's 16 query rows

  const ushort_t* Qb = Q  + (size_t)(b * H_ + h) * S_ * DH_;
  const ushort_t* Kb = K  + (size_t)(b * H_ + h) * S_ * DH_;
  const ushort_t* Vb = Vt + (size_t)(b * H_ + h) * DH_ * S_;

  // Q A-fragments for K-dim 0..31 and 32..63 (pairs contiguous along Dh)
  FragAB aq[2];
  {
    int m = qr0 + l16;
    #pragma unroll
    for (int st = 0; st < 2; ++st)
      #pragma unroll
      for (int r = 0; r < 8; ++r) {
        int k0 = st * 32 + hi * 16 + 2 * r;
        aq[st].u[r] = *reinterpret_cast<const uint32*>(Qb + (size_t)m * DH_ + k0);
      }
  }

  float mrow[8], lrow[8];
  #pragma unroll
  for (int i = 0; i < 8; ++i) { mrow[i] = -1e30f; lrow[i] = 0.f; }
  v8f zero8 = {0.f,0.f,0.f,0.f,0.f,0.f,0.f,0.f};
  v8f o[4] = { zero8, zero8, zero8, zero8 };

  ushort_t* Pw = &Plds[w][0];

  for (int j0 = 0; j0 <= qr0 + 15; j0 += 32) {       // causal per-wave bound (uniform)
    // ----- scores: two 16-key subtiles, each = 2 WMMAs over Dh -----
    FragC sc[2];
    #pragma unroll
    for (int t = 0; t < 2; ++t) {
      int j = j0 + t * 16 + l16;                      // key row = B column
      FragAB bk0, bk1;
      #pragma unroll
      for (int r = 0; r < 8; ++r) {
        int k = hi * 16 + 2 * r;                      // B-layout K pairs
        bk0.u[r] = *reinterpret_cast<const uint32*>(Kb + (size_t)j * DH_ + k);
        bk1.u[r] = *reinterpret_cast<const uint32*>(Kb + (size_t)j * DH_ + 32 + k);
      }
      v8f c = zero8;
      c = __builtin_amdgcn_wmma_f32_16x16x32_bf16(false, aq[0].v, false, bk0.v, (short)0, c, false, false);
      c = __builtin_amdgcn_wmma_f32_16x16x32_bf16(false, aq[1].v, false, bk1.v, (short)0, c, false, false);
      sc[t].v = c;
    }
    // ----- online softmax (base-2; row reductions across 16 lanes of each half) -----
    #pragma unroll
    for (int i = 0; i < 8; ++i) {
      int q    = qr0 + i + hi * 8;
      int key0 = j0 + l16;
      float s0 = sc[0].f[i] * SCALE_LOG2E;
      float s1 = sc[1].f[i] * SCALE_LOG2E;
      if (key0 > q)      s0 = -1e30f;
      if (key0 + 16 > q) s1 = -1e30f;
      float tmax = fmaxf(s0, s1);
      #pragma unroll
      for (int d = 1; d < 16; d <<= 1) tmax = fmaxf(tmax, __shfl_xor(tmax, d, 32));
      float mnew  = fmaxf(mrow[i], tmax);
      float alpha = exp2f(mrow[i] - mnew);
      float p0 = exp2f(s0 - mnew);
      float p1 = exp2f(s1 - mnew);
      float rs = p0 + p1;
      #pragma unroll
      for (int d = 1; d < 16; d <<= 1) rs += __shfl_xor(rs, d, 32);
      lrow[i] = lrow[i] * alpha + rs;
      mrow[i] = mnew;
      #pragma unroll
      for (int nt = 0; nt < 4; ++nt) o[nt][i] *= alpha;
      int prow = i + hi * 8;                          // C-layout row -> LDS
      Pw[prow * 34 + l16]      = f2bf(p0);
      Pw[prow * 34 + 16 + l16] = f2bf(p1);
    }
    // ----- P (16x32) as A-fragment via LDS transpose -----
    FragAB ap;
    #pragma unroll
    for (int r = 0; r < 8; ++r) {
      int k0 = hi * 16 + 2 * r;
      ap.u[r] = *reinterpret_cast<const uint32*>(&Pw[l16 * 34 + k0]);
    }
    // ----- O += P * V : 4 WMMAs over the 64-wide Dh (Vt pairs contiguous in S) -----
    #pragma unroll
    for (int nt = 0; nt < 4; ++nt) {
      FragAB bv;
      int e = nt * 16 + l16;
      #pragma unroll
      for (int r = 0; r < 8; ++r) {
        int k = hi * 16 + 2 * r;
        bv.u[r] = *reinterpret_cast<const uint32*>(Vb + (size_t)e * S_ + j0 + k);
      }
      o[nt] = __builtin_amdgcn_wmma_f32_16x16x32_bf16(false, ap.v, false, bv.v, (short)0, o[nt], false, false);
    }
  }

  // ----- normalize and write Z [B*S, H*Dh] bf16 -----
  #pragma unroll
  for (int nt = 0; nt < 4; ++nt) {
    FragC c; c.v = o[nt];
    #pragma unroll
    for (int i = 0; i < 8; ++i) {
      int q = qr0 + i + hi * 8;
      float inv = 1.0f / lrow[i];
      int col = h * DH_ + nt * 16 + l16;
      Z[(size_t)(b * S_ + q) * D_ + col] = f2bf(c.f[i] * inv);
    }
  }
}

// ---------------------------------------------------------------- launch
extern "C" void kernel_launch(void* const* d_in, const int* in_sizes, int n_in,
                              void* d_out, int out_size, void* d_ws, size_t ws_size,
                              hipStream_t stream) {
  (void)in_sizes; (void)n_in; (void)out_size; (void)ws_size;
  const float* residual = (const float*)d_in[0];
  const float* W_Q = (const float*)d_in[1];
  const float* W_K = (const float*)d_in[2];
  const float* W_V = (const float*)d_in[3];
  const float* W_O = (const float*)d_in[4];
  const float* b_Q = (const float*)d_in[5];
  const float* b_K = (const float*)d_in[6];
  const float* b_V = (const float*)d_in[7];
  const float* b_O = (const float*)d_in[8];

  char* ws = (char*)d_ws;
  size_t off = 0;
  auto wsAlloc = [&](size_t bytes) -> char* {
    char* p = ws + off; off += (bytes + 255) & ~(size_t)255; return p;
  };
  const size_t actBytes = (size_t)BS_ * D_ * sizeof(ushort_t);   // 12.58 MB each
  const size_t wBytes   = (size_t)D_ * D_ * sizeof(ushort_t);    // 1.18 MB each
  ushort_t* Xbf = (ushort_t*)wsAlloc(actBytes);
  ushort_t* Qbf = (ushort_t*)wsAlloc(actBytes);
  ushort_t* Kbf = (ushort_t*)wsAlloc(actBytes);
  ushort_t* Vtb = (ushort_t*)wsAlloc(actBytes);
  ushort_t* Zbf = (ushort_t*)wsAlloc(actBytes);
  ushort_t* WQt = (ushort_t*)wsAlloc(wBytes);
  ushort_t* WKt = (ushort_t*)wsAlloc(wBytes);
  ushort_t* WVt = (ushort_t*)wsAlloc(wBytes);
  ushort_t* WOt = (ushort_t*)wsAlloc(wBytes);

  const int nAct = BS_ * D_;
  const int nW   = D_ * D_;
  k_f32_to_bf16<<<(nAct + 255) / 256, 256, 0, stream>>>(residual, Xbf, nAct);
  k_pack_qkv<<<(nW + 255) / 256, 256, 0, stream>>>(W_Q, WQt);
  k_pack_qkv<<<(nW + 255) / 256, 256, 0, stream>>>(W_K, WKt);
  k_pack_qkv<<<(nW + 255) / 256, 256, 0, stream>>>(W_V, WVt);
  k_pack_oT<<<(nW + 255) / 256, 256, 0, stream>>>(W_O, WOt);

  dim3 gGemm(BS_ / GT_M, D_ / GT_N);   // 64 x 12
  k_gemm_bf16<<<gGemm, 256, 0, stream>>>(Xbf, WQt, b_Q, Qbf, 0);
  k_gemm_bf16<<<gGemm, 256, 0, stream>>>(Xbf, WKt, b_K, Kbf, 0);
  k_gemm_bf16<<<gGemm, 256, 0, stream>>>(Xbf, WVt, b_V, Vtb, 1);

  dim3 gAttn(S_ / 128, H_, B_);        // 16 x 12 x 4
  k_attn<<<gAttn, 256, 0, stream>>>(Qbf, Kbf, Vtb, Zbf);

  k_gemm_bf16<<<gGemm, 256, 0, stream>>>(Zbf, WOt, b_O, d_out, 2);
}
